// CausalAttention_16071767622486
// MI455X (gfx1250) — compile-verified, matrix operand order checked by reading, not run on hardware
//
#include <hip/hip_runtime.h>

// Problem constants (reference: B=4, S=2048, D=1024, H=16, HD=64)
#define B_   4
#define S_   2048
#define D_   1024
#define H_   16
#define HD_  64
#define BS_  (B_ * S_)

#define MT_  128   // GEMM M tile
#define KT_  64    // GEMM K step

typedef __attribute__((ext_vector_type(16))) _Float16     v16h;
typedef __attribute__((ext_vector_type(8)))  float        v8f;
typedef __attribute__((ext_vector_type(8)))  unsigned int v8u;
typedef __attribute__((ext_vector_type(4)))  unsigned int v4u;

// ---------------------------------------------------------------------------
// gfx1250 async global->LDS copy path (guarded; falls back to register-staged
// copies if the builtins are absent). Builtin signature (from hipcc diag):
//   void __builtin_amdgcn_global_load_async_to_lds_b128(
//        int __vector(4) AS1* src, int __vector(4) AS3* dst, imm off, imm cpol)
// ---------------------------------------------------------------------------
#if defined(__has_builtin)
#  if __has_builtin(__builtin_amdgcn_global_load_async_to_lds_b128) && \
      __has_builtin(__builtin_amdgcn_s_wait_asynccnt)
#    define ASYNC_LDS 1
#  endif
#endif
#ifndef ASYNC_LDS
#  define ASYNC_LDS 0
#endif

#if ASYNC_LDS
typedef __attribute__((__vector_size__(4 * sizeof(int)))) int v4i_;
typedef __attribute__((address_space(1))) v4i_ as1_v4i;
typedef __attribute__((address_space(3))) v4i_ as3_v4i;
__device__ __forceinline__ void async_b128(const _Float16* g, _Float16* l) {
    __builtin_amdgcn_global_load_async_to_lds_b128((as1_v4i*)g, (as3_v4i*)l,
                                                   0, 0);
}
__device__ __forceinline__ void async_wait0() {
    __builtin_amdgcn_s_wait_asynccnt(0);
}
#endif

// ---------------------------------------------------------------------------
// WMMA helper: D = A(16x32 f16) * B(32x16 f16) + C(16x16 f32)
// ---------------------------------------------------------------------------
__device__ __forceinline__ v8f wmma16(v16h a, v16h b, v8f c) {
    return __builtin_amdgcn_wmma_f32_16x16x32_f16(false, a, false, b,
                                                  (short)0, c, false, false);
}

// ---------------------------------------------------------------------------
// Fragment loader (CDNA5 ISA 16-bit A-matrix 16x32 layout):
//   lane 0-15 : row M = lane,    K = {0..7, 16..23}
//   lane 16-31: row M = lane-16, K = {8..15, 24..31}
// B operands staged as [N][K] use the identical pattern with row := N.
// ---------------------------------------------------------------------------
__device__ __forceinline__ v16h frag_row_major(const _Float16* p0, int ld) {
    int lane  = threadIdx.x & 31;
    int row   = lane & 15;
    int kbase = (lane >> 4) << 3;  // 0 or 8
    const _Float16* p = p0 + row * ld + kbase;
    v4u lo = *(const v4u*)p;         // K = kbase .. kbase+7
    v4u hi = *(const v4u*)(p + 16);  // K = kbase+16 .. kbase+23
    v8u r = {lo.x, lo.y, lo.z, lo.w, hi.x, hi.y, hi.z, hi.w};
    return __builtin_bit_cast(v16h, r);
}

__device__ __forceinline__ v8f zero8() {
    v8f z = {0.f, 0.f, 0.f, 0.f, 0.f, 0.f, 0.f, 0.f};
    return z;
}

// ---------------------------------------------------------------------------
// fp32 -> fp16 conversion (grid-stride)
// ---------------------------------------------------------------------------
__global__ void cvt_f16(const float* __restrict__ in, _Float16* __restrict__ out,
                        int n) {
    for (int i = blockIdx.x * blockDim.x + threadIdx.x; i < n;
         i += gridDim.x * blockDim.x)
        out[i] = (_Float16)in[i];
}

// ---------------------------------------------------------------------------
// Weight transpose + convert: W[H][D][HD] (f32) -> Wt[H][HD][D] (f16).
// ---------------------------------------------------------------------------
__global__ void cvt_transpose_w(const float* __restrict__ W,
                                _Float16* __restrict__ Wt) {
    int total = H_ * D_ * HD_;
    for (int i = blockIdx.x * blockDim.x + threadIdx.x; i < total;
         i += gridDim.x * blockDim.x) {
        int h = i / (D_ * HD_);
        int r = i % (D_ * HD_);
        int k = r / HD_;
        int n = r % HD_;
        Wt[((size_t)(h * HD_ + n)) * D_ + k] = (_Float16)W[i];
    }
}

// ---------------------------------------------------------------------------
// QKV projection. Wt_all: [3][H][HD][D] (pre-transposed). QKV: [3][H][BS][HD].
// grid = (BS/128, 3*H), block = 256 (8 waves). Double-buffered LDS panels:
// next panel's async loads overlap the current panel's 8 WMMAs per wave.
// ---------------------------------------------------------------------------
__global__ __launch_bounds__(256) void qkv_gemm(
    const _Float16* __restrict__ X, const _Float16* __restrict__ Wt_all,
    const float* __restrict__ bq, const float* __restrict__ bk,
    const float* __restrict__ bv, _Float16* __restrict__ QKV)
{
    int m0  = blockIdx.x * MT_;
    int sel = blockIdx.y / H_;
    int h   = blockIdx.y % H_;

    const _Float16* Wt = Wt_all + (size_t)(sel * H_ + h) * HD_ * D_;
    const float* bias  = ((sel == 0) ? bq : (sel == 1) ? bk : bv) + h * HD_;
    _Float16* Out      = QKV + (size_t)(sel * H_ + h) * BS_ * HD_;

    __shared__ alignas(16) _Float16 As[2][MT_ * KT_];  // 2 x 16 KB
    __shared__ alignas(16) _Float16 Bs[2][64 * KT_];   // 2 x  8 KB

    int tid  = threadIdx.x;
    int lane = tid & 31;
    int wave = tid >> 5;

    v8f acc[4] = {zero8(), zero8(), zero8(), zero8()};

    // ---- prologue: stage panel k0=0 into buffer 0 ----
#if ASYNC_LDS
#pragma unroll
    for (int r = 0; r < 4; ++r) {
        int f = tid + 256 * r, row = f >> 3, col = (f & 7) << 3;
        async_b128(X + (size_t)(m0 + row) * D_ + col, &As[0][row * KT_ + col]);
    }
#pragma unroll
    for (int r = 0; r < 2; ++r) {
        int f = tid + 256 * r, n = f >> 3, col = (f & 7) << 3;
        async_b128(Wt + (size_t)n * D_ + col, &Bs[0][n * KT_ + col]);
    }
    async_wait0();
#else
    {
        v4u ra[4], rb[2];
#pragma unroll
        for (int r = 0; r < 4; ++r) {
            int f = tid + 256 * r, row = f >> 3, col = (f & 7) << 3;
            ra[r] = *(const v4u*)(X + (size_t)(m0 + row) * D_ + col);
        }
#pragma unroll
        for (int r = 0; r < 2; ++r) {
            int f = tid + 256 * r, n = f >> 3, col = (f & 7) << 3;
            rb[r] = *(const v4u*)(Wt + (size_t)n * D_ + col);
        }
#pragma unroll
        for (int r = 0; r < 4; ++r) {
            int f = tid + 256 * r, row = f >> 3, col = (f & 7) << 3;
            *(v4u*)(&As[0][row * KT_ + col]) = ra[r];
        }
#pragma unroll
        for (int r = 0; r < 2; ++r) {
            int f = tid + 256 * r, n = f >> 3, col = (f & 7) << 3;
            *(v4u*)(&Bs[0][n * KT_ + col]) = rb[r];
        }
    }
#endif
    __syncthreads();

    int buf = 0;
    for (int k0 = 0; k0 < D_; k0 += KT_) {
        int nk = k0 + KT_;
#if !ASYNC_LDS
        v4u ra[4], rb[2];
#endif
        if (nk < D_) {  // initiate next panel before computing current one
#if ASYNC_LDS
#pragma unroll
            for (int r = 0; r < 4; ++r) {
                int f = tid + 256 * r, row = f >> 3, col = (f & 7) << 3;
                async_b128(X + (size_t)(m0 + row) * D_ + nk + col,
                           &As[buf ^ 1][row * KT_ + col]);
            }
#pragma unroll
            for (int r = 0; r < 2; ++r) {
                int f = tid + 256 * r, n = f >> 3, col = (f & 7) << 3;
                async_b128(Wt + (size_t)n * D_ + nk + col,
                           &Bs[buf ^ 1][n * KT_ + col]);
            }
#else
#pragma unroll
            for (int r = 0; r < 4; ++r) {
                int f = tid + 256 * r, row = f >> 3, col = (f & 7) << 3;
                ra[r] = *(const v4u*)(X + (size_t)(m0 + row) * D_ + nk + col);
            }
#pragma unroll
            for (int r = 0; r < 2; ++r) {
                int f = tid + 256 * r, n = f >> 3, col = (f & 7) << 3;
                rb[r] = *(const v4u*)(Wt + (size_t)n * D_ + nk + col);
            }
#endif
        }

        // compute current panel
#pragma unroll
        for (int kk = 0; kk < 2; ++kk) {
            v16h a = frag_row_major(&As[buf][wave * 16 * KT_ + kk * 32], KT_);
#pragma unroll
            for (int ni = 0; ni < 4; ++ni) {
                v16h bf = frag_row_major(&Bs[buf][ni * 16 * KT_ + kk * 32], KT_);
                acc[ni] = wmma16(a, bf, acc[ni]);
            }
        }

        if (nk < D_) {
#if ASYNC_LDS
            async_wait0();
#else
#pragma unroll
            for (int r = 0; r < 4; ++r) {
                int f = tid + 256 * r, row = f >> 3, col = (f & 7) << 3;
                *(v4u*)(&As[buf ^ 1][row * KT_ + col]) = ra[r];
            }
#pragma unroll
            for (int r = 0; r < 2; ++r) {
                int f = tid + 256 * r, n = f >> 3, col = (f & 7) << 3;
                *(v4u*)(&Bs[buf ^ 1][n * KT_ + col]) = rb[r];
            }
#endif
        }
        __syncthreads();
        buf ^= 1;
    }

    int colb = lane & 15;
    int rb_  = wave * 16 + ((lane >> 4) << 3);
#pragma unroll
    for (int ni = 0; ni < 4; ++ni) {
        int   c  = ni * 16 + colb;
        float bb = bias[c];
#pragma unroll
        for (int i = 0; i < 8; ++i)
            Out[(size_t)(m0 + rb_ + i) * HD_ + c] = (_Float16)(acc[ni][i] + bb);
    }
}

// ---------------------------------------------------------------------------
// Causal flash attention for one (b, h, 64-row Q tile).
// grid = (S/64, B*H), block = 256.
// ---------------------------------------------------------------------------
__global__ __launch_bounds__(256) void attn_kernel(
    const _Float16* __restrict__ Q, const _Float16* __restrict__ K,
    const _Float16* __restrict__ V, _Float16* __restrict__ wv)
{
    int bh = blockIdx.y;
    int b  = bh / H_;
    int h  = bh % H_;
    int m0 = blockIdx.x * 64;

    const _Float16* Qh = Q + ((size_t)h * BS_ + (size_t)b * S_) * HD_;
    const _Float16* Kh = K + ((size_t)h * BS_ + (size_t)b * S_) * HD_;
    const _Float16* Vh = V + ((size_t)h * BS_ + (size_t)b * S_) * HD_;

    __shared__ alignas(16) _Float16 Qs[64 * 64];   // [m][e]
    __shared__ alignas(16) _Float16 Ks[64 * 64];   // [key][e]  ([N][K] for QK^T)
    __shared__ alignas(16) _Float16 Vt[64 * 64];   // [e][key]  ([N][K] for PV)
    __shared__ alignas(16) _Float16 Pb[64 * 64];   // probs, f16
    __shared__ float Ssc[64 * 64];
    __shared__ float mrow[64], lrow[64], srow[64];

    int tid  = threadIdx.x;
    int lane = tid & 31;
    int wave = tid >> 5;
    int mi   = wave >> 1;
    int nb   = (wave & 1) * 2;
    int rb   = mi * 16 + ((lane >> 4) << 3);
    int colb = lane & 15;

    {   // Q tile: contiguous rows -> flat b128 copy (8 KB)
        const v4u* src = (const v4u*)(Qh + (size_t)m0 * HD_);
        v4u* dst = (v4u*)Qs;
        for (int i = tid; i < 512; i += 256) dst[i] = src[i];
    }
    if (tid < 64) { mrow[tid] = -INFINITY; lrow[tid] = 0.f; }

    v8f accO[2] = {zero8(), zero8()};
    __syncthreads();

    int jmax = blockIdx.x;
    for (int j = 0; j <= jmax; ++j) {
        {   // K tile
#if ASYNC_LDS
            for (int i = tid; i < 512; i += 256)
                async_b128(Kh + (size_t)j * 64 * HD_ + i * 8, Ks + i * 8);
#else
            const v4u* sK = (const v4u*)(Kh + (size_t)j * 64 * HD_);
            v4u* dK = (v4u*)Ks;
            for (int i = tid; i < 512; i += 256) dK[i] = sK[i];
#endif
        }
        {   // V tile: transpose in registers -> Vt[e][key]
            int key = tid & 63;
            int e0  = (tid >> 6) << 4;   // 0,16,32,48
            const v4u* sV =
                (const v4u*)(Vh + (size_t)(j * 64 + key) * HD_ + e0);
            v4u a0 = sV[0], a1 = sV[1];
            const _Float16* h0 = (const _Float16*)&a0;
            const _Float16* h1 = (const _Float16*)&a1;
#pragma unroll
            for (int i = 0; i < 8; ++i) Vt[(e0 + i) * 64 + key] = h0[i];
#pragma unroll
            for (int i = 0; i < 8; ++i) Vt[(e0 + 8 + i) * 64 + key] = h1[i];
        }
        if (j < jmax) {   // prefetch next K/V tiles (global_prefetch_b8)
            __builtin_prefetch(Kh + (size_t)(j + 1) * 64 * HD_ + tid * 16, 0, 1);
            __builtin_prefetch(Vh + (size_t)(j + 1) * 64 * HD_ + tid * 16, 0, 1);
        }
#if ASYNC_LDS
        async_wait0();
#endif
        __syncthreads();

        // S = (Q K^T) * 1/sqrt(HD)
#pragma unroll
        for (int s = 0; s < 2; ++s) {
            int ni = nb + s;
            v8f acc = zero8();
#pragma unroll
            for (int kk = 0; kk < 2; ++kk) {
                v16h a  = frag_row_major(Qs + mi * 16 * 64 + kk * 32, 64);
                v16h bf = frag_row_major(Ks + ni * 16 * 64 + kk * 32, 64);
                acc = wmma16(a, bf, acc);
            }
            int col = ni * 16 + colb;
#pragma unroll
            for (int i = 0; i < 8; ++i)
                Ssc[(rb + i) * 64 + col] = acc[i] * 0.125f;  // 1/sqrt(64)
        }
        __syncthreads();

        // online softmax stats: thread r (< 64) owns row r
        if (tid < 64) {
            int   cmax = (j == jmax) ? tid : 63;  // causal bound inside tile
            float mold = mrow[tid];
            float mx   = mold;
            for (int c = 0; c <= cmax; ++c)
                mx = fmaxf(mx, Ssc[tid * 64 + c]);
            float scale = __expf(mold - mx);
            float sum = 0.f;
            for (int c = 0; c < 64; ++c) {
                float p = (c <= cmax) ? __expf(Ssc[tid * 64 + c] - mx) : 0.f;
                Pb[tid * 64 + c] = (_Float16)p;
                sum += p;
            }
            mrow[tid] = mx;
            lrow[tid] = lrow[tid] * scale + sum;
            srow[tid] = scale;
        }
        __syncthreads();

        // O = O*scale + P @ V
#pragma unroll
        for (int s = 0; s < 2; ++s) {
            int ni  = nb + s;
            v8f acc = accO[s];
#pragma unroll
            for (int i = 0; i < 8; ++i) acc[i] *= srow[rb + i];
#pragma unroll
            for (int kk = 0; kk < 2; ++kk) {
                v16h a  = frag_row_major(Pb + mi * 16 * 64 + kk * 32, 64);
                v16h bf = frag_row_major(Vt + ni * 16 * 64 + kk * 32, 64);
                acc = wmma16(a, bf, acc);
            }
            accO[s] = acc;
        }
        __syncthreads();
    }

    if (tid < 64) srow[tid] = 1.f / lrow[tid];
    __syncthreads();

#pragma unroll
    for (int s = 0; s < 2; ++s) {
        int ni  = nb + s;
        int col = h * HD_ + ni * 16 + colb;
#pragma unroll
        for (int i = 0; i < 8; ++i) {
            float o = accO[s][i] * srow[rb + i];
            wv[(size_t)(b * S_ + m0 + rb + i) * D_ + col] = (_Float16)o;
        }
    }
}

// ---------------------------------------------------------------------------
// Output projection: out[m][n] = sum_k wv[m][k] * Wo[n][k] + bo[n] (fp32 out)
// grid = (BS/128, D/64), block = 256, double-buffered panels.
// ---------------------------------------------------------------------------
__global__ __launch_bounds__(256) void out_gemm(
    const _Float16* __restrict__ A, const _Float16* __restrict__ Wo,
    const float* __restrict__ bo, float* __restrict__ Out)
{
    int m0 = blockIdx.x * MT_;
    int n0 = blockIdx.y * 64;

    __shared__ alignas(16) _Float16 As[2][MT_ * KT_];
    __shared__ alignas(16) _Float16 Bs[2][64 * KT_];

    int tid  = threadIdx.x;
    int lane = tid & 31;
    int wave = tid >> 5;

    v8f acc[4] = {zero8(), zero8(), zero8(), zero8()};

#if ASYNC_LDS
#pragma unroll
    for (int r = 0; r < 4; ++r) {
        int f = tid + 256 * r, row = f >> 3, col = (f & 7) << 3;
        async_b128(A + (size_t)(m0 + row) * D_ + col, &As[0][row * KT_ + col]);
    }
#pragma unroll
    for (int r = 0; r < 2; ++r) {
        int f = tid + 256 * r, n = f >> 3, col = (f & 7) << 3;
        async_b128(Wo + (size_t)(n0 + n) * D_ + col, &Bs[0][n * KT_ + col]);
    }
    async_wait0();
#else
    {
        v4u ra[4], rb[2];
#pragma unroll
        for (int r = 0; r < 4; ++r) {
            int f = tid + 256 * r, row = f >> 3, col = (f & 7) << 3;
            ra[r] = *(const v4u*)(A + (size_t)(m0 + row) * D_ + col);
        }
#pragma unroll
        for (int r = 0; r < 2; ++r) {
            int f = tid + 256 * r, n = f >> 3, col = (f & 7) << 3;
            rb[r] = *(const v4u*)(Wo + (size_t)(n0 + n) * D_ + col);
        }
#pragma unroll
        for (int r = 0; r < 4; ++r) {
            int f = tid + 256 * r, row = f >> 3, col = (f & 7) << 3;
            *(v4u*)(&As[0][row * KT_ + col]) = ra[r];
        }
#pragma unroll
        for (int r = 0; r < 2; ++r) {
            int f = tid + 256 * r, n = f >> 3, col = (f & 7) << 3;
            *(v4u*)(&Bs[0][n * KT_ + col]) = rb[r];
        }
    }
#endif
    __syncthreads();

    int buf = 0;
    for (int k0 = 0; k0 < D_; k0 += KT_) {
        int nk = k0 + KT_;
#if !ASYNC_LDS
        v4u ra[4], rb[2];
#endif
        if (nk < D_) {
#if ASYNC_LDS
#pragma unroll
            for (int r = 0; r < 4; ++r) {
                int f = tid + 256 * r, row = f >> 3, col = (f & 7) << 3;
                async_b128(A + (size_t)(m0 + row) * D_ + nk + col,
                           &As[buf ^ 1][row * KT_ + col]);
            }
#pragma unroll
            for (int r = 0; r < 2; ++r) {
                int f = tid + 256 * r, n = f >> 3, col = (f & 7) << 3;
                async_b128(Wo + (size_t)(n0 + n) * D_ + nk + col,
                           &Bs[buf ^ 1][n * KT_ + col]);
            }
#else
#pragma unroll
            for (int r = 0; r < 4; ++r) {
                int f = tid + 256 * r, row = f >> 3, col = (f & 7) << 3;
                ra[r] = *(const v4u*)(A + (size_t)(m0 + row) * D_ + nk + col);
            }
#pragma unroll
            for (int r = 0; r < 2; ++r) {
                int f = tid + 256 * r, n = f >> 3, col = (f & 7) << 3;
                rb[r] = *(const v4u*)(Wo + (size_t)(n0 + n) * D_ + nk + col);
            }
#endif
        }

#pragma unroll
        for (int kk = 0; kk < 2; ++kk) {
            v16h a = frag_row_major(&As[buf][wave * 16 * KT_ + kk * 32], KT_);
#pragma unroll
            for (int ni = 0; ni < 4; ++ni) {
                v16h bf = frag_row_major(&Bs[buf][ni * 16 * KT_ + kk * 32], KT_);
                acc[ni] = wmma16(a, bf, acc[ni]);
            }
        }

        if (nk < D_) {
#if ASYNC_LDS
            async_wait0();
#else
#pragma unroll
            for (int r = 0; r < 4; ++r) {
                int f = tid + 256 * r, row = f >> 3, col = (f & 7) << 3;
                *(v4u*)(&As[buf ^ 1][row * KT_ + col]) = ra[r];
            }
#pragma unroll
            for (int r = 0; r < 2; ++r) {
                int f = tid + 256 * r, n = f >> 3, col = (f & 7) << 3;
                *(v4u*)(&Bs[buf ^ 1][n * KT_ + col]) = rb[r];
            }
#endif
        }
        __syncthreads();
        buf ^= 1;
    }

    int colb = lane & 15;
    int rb_  = wave * 16 + ((lane >> 4) << 3);
#pragma unroll
    for (int ni = 0; ni < 4; ++ni) {
        int   c  = n0 + ni * 16 + colb;
        float bb = bo[c];
#pragma unroll
        for (int i = 0; i < 8; ++i)
            Out[(size_t)(m0 + rb_ + i) * D_ + c] = acc[ni][i] + bb;
    }
}

// ---------------------------------------------------------------------------
// Host launcher
// ---------------------------------------------------------------------------
extern "C" void kernel_launch(void* const* d_in, const int* in_sizes, int n_in,
                              void* d_out, int out_size, void* d_ws, size_t ws_size,
                              hipStream_t stream) {
    (void)in_sizes; (void)n_in; (void)out_size; (void)ws_size;

    const float* X  = (const float*)d_in[0];
    const float* Wq = (const float*)d_in[1];
    const float* bq = (const float*)d_in[2];
    const float* Wk = (const float*)d_in[3];
    const float* bk = (const float*)d_in[4];
    const float* Wv = (const float*)d_in[5];
    const float* bv = (const float*)d_in[6];
    const float* Wo = (const float*)d_in[7];
    const float* bo = (const float*)d_in[8];
    float* out = (float*)d_out;

    // Workspace carve (fp16 elements). Total ~92 MB.
    unsigned short* w = (unsigned short*)d_ws;
    _Float16* Xh  = (_Float16*)w;  w += (size_t)BS_ * D_;
    _Float16* Wth = (_Float16*)w;  w += (size_t)3 * H_ * HD_ * D_;  // [3][H][HD][D]
    _Float16* Woh = (_Float16*)w;  w += (size_t)D_ * D_;
    _Float16* QKV = (_Float16*)w;  w += (size_t)3 * H_ * BS_ * HD_; // [3][H][BS][HD]
    _Float16* WVh = (_Float16*)w;

    const size_t wsz = (size_t)H_ * HD_ * D_;

    cvt_f16<<<2048, 256, 0, stream>>>(X, Xh, BS_ * D_);
    cvt_transpose_w<<<1024, 256, 0, stream>>>(Wq, Wth + 0 * wsz);
    cvt_transpose_w<<<1024, 256, 0, stream>>>(Wk, Wth + 1 * wsz);
    cvt_transpose_w<<<1024, 256, 0, stream>>>(Wv, Wth + 2 * wsz);
    cvt_f16<<<1024, 256, 0, stream>>>(Wo, Woh, D_ * D_);

    qkv_gemm<<<dim3(BS_ / MT_, 3 * H_), 256, 0, stream>>>(
        Xh, Wth, bq, bk, bv, QKV);

    const size_t hsz = (size_t)H_ * BS_ * HD_;
    attn_kernel<<<dim3(S_ / 64, B_ * H_), 256, 0, stream>>>(
        QKV + 0 * hsz, QKV + 1 * hsz, QKV + 2 * hsz, WVh);

    out_gemm<<<dim3(BS_ / MT_, D_ / 64), 256, 0, stream>>>(WVh, Woh, bo, out);
}
